// Shakespeare_88338887344717
// MI455X (gfx1250) — matrix operand order repaired and checked by reading, not compiled, not run on hardware
//
#include <hip/hip_runtime.h>

#define DEV_INLINE __device__ __forceinline__

typedef __bf16 bf16_t;
typedef bf16_t v16bf __attribute__((ext_vector_type(16)));
typedef float  v8f   __attribute__((ext_vector_type(8)));
typedef unsigned int u32x4 __attribute__((ext_vector_type(4)));
typedef int v4i __attribute__((ext_vector_type(4)));

constexpr int Vv = 32000, Ee = 768, Hh = 12, Ll = 12, Tt = 1024, Bb = 4, Dd = 64;
constexpr int Mr = Bb * Tt;   // 4096 token rows
constexpr int Ff = 4 * Ee;    // 3072

union Frag { u32x4 q[2]; v16bf v; unsigned short u[16]; };
union AccF { v8f v; float f[8]; };

DEV_INLINE unsigned short f2bf(float x) {
  union { float f; unsigned u; } c; c.f = x;
  unsigned r = c.u + 0x7FFFu + ((c.u >> 16) & 1u);   // round-to-nearest-even
  return (unsigned short)(r >> 16);
}

DEV_INLINE v8f wmma_bf16(v16bf a, v16bf b, v8f c) {
  return __builtin_amdgcn_wmma_f32_16x16x32_bf16(false, a, false, b, (short)0, c, false, false);
}

// CDNA5 async global->LDS path (ASYNCcnt-tracked), guarded so the file still
// builds if the toolchain lacks the builtin.
#if defined(__gfx1250__) && __has_builtin(__builtin_amdgcn_global_load_async_to_lds_b128)
#define USE_ASYNC_LDS 1
#define AS1 __attribute__((address_space(1)))
#define AS3 __attribute__((address_space(3)))
DEV_INLINE void async_wait0() {
#if __has_builtin(__builtin_amdgcn_s_wait_asynccnt)
  __builtin_amdgcn_s_wait_asynccnt(0);
#else
  asm volatile("s_wait_asynccnt 0x0" ::: "memory");
#endif
}
#endif

// ---------------- elementwise f32 -> bf16 convert ----------------
__global__ __launch_bounds__(256) void cvt_kernel(const float* __restrict__ in,
                                                  unsigned short* __restrict__ out,
                                                  size_t n) {
  size_t i = (size_t)blockIdx.x * 256 + threadIdx.x;
  if (i < n) out[i] = f2bf(in[i]);
}

// ---------------- embedding: x = tok_emb[tokens] + pos_emb ----------------
__global__ __launch_bounds__(256) void embed_kernel(const float* __restrict__ tok,
                                                    const float* __restrict__ pos,
                                                    const int* __restrict__ tokens,
                                                    float* __restrict__ x) {
  int row = blockIdx.x;               // 0..Mr-1
  int t = row % Tt;
  int id = tokens[row];
  const float* tr = tok + (size_t)id * Ee;
  const float* pr = pos + (size_t)t * Ee;
  float* xr = x + (size_t)row * Ee;
  for (int c = threadIdx.x; c < Ee; c += 256) xr[c] = tr[c] + pr[c];
}

// ---------------- LayerNorm (f32 in) -> bf16 out ----------------
__global__ __launch_bounds__(256) void ln_kernel(const float* __restrict__ x,
                                                 const float* __restrict__ g,
                                                 const float* __restrict__ b,
                                                 unsigned short* __restrict__ out) {
  int row = blockIdx.x;
  __shared__ float s1[256], s2[256];
  const float* xr = x + (size_t)row * Ee;
  float a = 0.f, q = 0.f;
  for (int c = threadIdx.x; c < Ee; c += 256) { float v = xr[c]; a += v; q += v * v; }
  s1[threadIdx.x] = a; s2[threadIdx.x] = q; __syncthreads();
  for (int o = 128; o > 0; o >>= 1) {
    if (threadIdx.x < (unsigned)o) { s1[threadIdx.x] += s1[threadIdx.x + o]; s2[threadIdx.x] += s2[threadIdx.x + o]; }
    __syncthreads();
  }
  float mu = s1[0] * (1.0f / Ee);
  float var = s2[0] * (1.0f / Ee) - mu * mu;
  float rs = rsqrtf(var + 1e-5f);
  unsigned short* orow = out + (size_t)row * Ee;
  for (int c = threadIdx.x; c < Ee; c += 256)
    orow[c] = f2bf((xr[c] - mu) * rs * g[c] + b[c]);
}

// ---------------- tiled bf16 WMMA GEMM ----------------
// C[Mr x Nn] = A[Mr x Kk](bf16,row-major) @ B[Kk x Nn](bf16,row-major) (+bias)(+epi)
// Block tile 128x64 (8 waves, 4x2), wave tile 32x32 -> 4 WMMAs per K-step.
// A tile staged via CDNA5 async global->LDS (b128); B tile transposed manually.
// EPI: 0 = f32 out, 1 = bf16 out, 2 = gelu->bf16 out, 3 = f32 residual add in place
template <int EPI, bool BIAS>
__global__ __launch_bounds__(256) void gemm_kernel(const unsigned short* __restrict__ A,
                                                   const unsigned short* __restrict__ Bm,
                                                   const float* __restrict__ bias,
                                                   float* __restrict__ Cf,
                                                   unsigned short* __restrict__ Cb,
                                                   int Nn, int Kk) {
  __shared__ __align__(16) unsigned short As[128][32];
  __shared__ __align__(16) unsigned short Bs[64][40];   // 80B rows keep 16B-aligned frags
  const int bm = blockIdx.x, bn = blockIdx.y;
  const int tid = threadIdx.x;
  const int w = tid >> 5, l = tid & 31;
  const int lh = l >> 4, ln = l & 15;
  const int wm = w >> 1, wn = w & 1;                    // 4x2 wave grid

  AccF acc[2][2];
  for (int i = 0; i < 2; ++i)
    for (int j = 0; j < 2; ++j)
      for (int r = 0; r < 8; ++r) acc[i][j].f[r] = 0.f;

  const int arow = tid >> 1, acol = (tid & 1) * 16;     // A: 128 rows x 32 cols, 2x16B/thread
  const int bk = tid >> 3,  bn8 = (tid & 7) * 8;        // B: 32 k-rows x 64 n-cols

  const unsigned short* aRow = A + (size_t)(bm * 128 + arow) * Kk + acol;
  const unsigned short* bRow = Bm + (size_t)bk * Nn + bn * 64 + bn8;

  for (int k0 = 0; k0 < Kk; k0 += 32) {
    // prefetch next K-tile (global_prefetch_b8)
    if (k0 + 32 < Kk) {
      __builtin_prefetch((const void*)(aRow + k0 + 32), 0, 1);
      __builtin_prefetch((const void*)(bRow + (size_t)(k0 + 32) * Nn), 0, 1);
    }
#ifdef USE_ASYNC_LDS
    __builtin_amdgcn_global_load_async_to_lds_b128(
        (AS1 v4i*)(aRow + k0), (AS3 v4i*)&As[arow][acol], 0, 0);
    __builtin_amdgcn_global_load_async_to_lds_b128(
        (AS1 v4i*)(aRow + k0 + 8), (AS3 v4i*)&As[arow][acol + 8], 0, 0);
#else
    *(u32x4*)&As[arow][acol]     = *(const u32x4*)(aRow + k0);
    *(u32x4*)&As[arow][acol + 8] = *(const u32x4*)(aRow + k0 + 8);
#endif
    // B tile: coalesced global read, transposed store into [n][k]
    u32x4 bv = *(const u32x4*)(bRow + (size_t)k0 * Nn);
    const unsigned short* bs = (const unsigned short*)&bv;
    #pragma unroll
    for (int i = 0; i < 8; ++i) Bs[bn8 + i][bk] = bs[i];
#ifdef USE_ASYNC_LDS
    async_wait0();
#endif
    __syncthreads();

    Frag fa[2], fb[2];
    #pragma unroll
    for (int i = 0; i < 2; ++i) {
      fa[i].q[0] = *(const u32x4*)&As[wm * 32 + i * 16 + ln][lh * 8];
      fa[i].q[1] = *(const u32x4*)&As[wm * 32 + i * 16 + ln][16 + lh * 8];
    }
    #pragma unroll
    for (int j = 0; j < 2; ++j) {
      fb[j].q[0] = *(const u32x4*)&Bs[wn * 32 + j * 16 + ln][lh * 8];
      fb[j].q[1] = *(const u32x4*)&Bs[wn * 32 + j * 16 + ln][16 + lh * 8];
    }
    #pragma unroll
    for (int i = 0; i < 2; ++i)
      #pragma unroll
      for (int j = 0; j < 2; ++j)
        acc[i][j].v = wmma_bf16(fa[i].v, fb[j].v, acc[i][j].v);
    __syncthreads();
  }

  for (int i = 0; i < 2; ++i) {
    for (int j = 0; j < 2; ++j) {
      for (int r = 0; r < 8; ++r) {
        int gm = bm * 128 + wm * 32 + i * 16 + r + lh * 8;
        int gn = bn * 64 + wn * 32 + j * 16 + ln;
        float v = acc[i][j].f[r];
        if (BIAS) v += bias[gn];
        size_t idx = (size_t)gm * Nn + gn;
        if (EPI == 0)      Cf[idx] = v;
        else if (EPI == 1) Cb[idx] = f2bf(v);
        else if (EPI == 2) Cb[idx] = f2bf(0.5f * v * (1.0f + erff(v * 0.70710678118f)));
        else               Cf[idx] = Cf[idx] + v;   // residual add, elementwise in place
      }
    }
  }
}

// ---------------- fused flash attention (causal), one wave per 16 q-rows ----------------
__global__ __launch_bounds__(32) void attn_kernel(const unsigned short* __restrict__ Q,
                                                  const unsigned short* __restrict__ Kp,
                                                  const unsigned short* __restrict__ Vp,
                                                  unsigned short* __restrict__ O) {
  const int qt = blockIdx.x, h = blockIdx.y, b = blockIdx.z;
  const int l = threadIdx.x, lh = l >> 4, ln = l & 15;
  __shared__ __align__(16) unsigned short sP[16][32];

  // Q tile 16x64 as two A-frags (d-chunks of 32)
  Frag qa[2];
  {
    const unsigned short* qb_ = Q + ((size_t)(b * Tt) + qt * 16 + ln) * Ee + h * Dd;
    for (int c = 0; c < 2; ++c) {
      qa[c].q[0] = *(const u32x4*)(qb_ + c * 32 + lh * 8);
      qa[c].q[1] = *(const u32x4*)(qb_ + c * 32 + 16 + lh * 8);
    }
  }

  float mx[8], sm[8];
  AccF oacc[4];
  for (int r = 0; r < 8; ++r) {
    mx[r] = -__builtin_inff(); sm[r] = 0.f;
    for (int jd = 0; jd < 4; ++jd) oacc[jd].f[r] = 0.f;
  }
  const float scale = 0.036084391824351615f;  // 1/sqrt(E)  (faithful to reference)
  const int qmax = qt * 16 + 15;

  for (int s0 = 0; s0 <= qmax; s0 += 32) {
    // S = Q @ K^T : two 16x16 D-tiles
    AccF sacc[2];
    for (int j = 0; j < 2; ++j) for (int r = 0; r < 8; ++r) sacc[j].f[r] = 0.f;
    for (int j = 0; j < 2; ++j) {
      int srow = s0 + j * 16 + ln; if (srow >= Tt) srow = Tt - 1;
      const unsigned short* kb = Kp + ((size_t)(b * Tt) + srow) * Ee + h * Dd;
      for (int c = 0; c < 2; ++c) {
        Frag f;
        f.q[0] = *(const u32x4*)(kb + c * 32 + lh * 8);
        f.q[1] = *(const u32x4*)(kb + c * 32 + 16 + lh * 8);
        sacc[j].v = wmma_bf16(qa[c].v, f.v, sacc[j].v);
      }
    }
    // online softmax (each C-row lives across 16 lanes of a half-wave)
    for (int r = 0; r < 8; ++r) {
      const int qi = qt * 16 + r + lh * 8;
      float s0v = sacc[0].f[r] * scale;
      float s1v = sacc[1].f[r] * scale;
      if (s0 + ln > qi)      s0v = -__builtin_inff();
      if (s0 + 16 + ln > qi) s1v = -__builtin_inff();
      float tmax = fmaxf(s0v, s1v);
      for (int o = 1; o < 16; o <<= 1) tmax = fmaxf(tmax, __shfl_xor(tmax, o, 32));
      float mn = fmaxf(mx[r], tmax);
      float corr = __expf(mx[r] - mn);
      mx[r] = mn;
      float p0 = __expf(s0v - mn), p1 = __expf(s1v - mn);
      float ps = p0 + p1;
      for (int o = 1; o < 16; o <<= 1) ps += __shfl_xor(ps, o, 32);
      sm[r] = sm[r] * corr + ps;
      for (int jd = 0; jd < 4; ++jd) oacc[jd].f[r] *= corr;
      sacc[0].f[r] = p0; sacc[1].f[r] = p1;
    }
    // re-layout P: D-layout -> A-layout through 1KB LDS
    for (int r = 0; r < 8; ++r) {
      sP[r + lh * 8][ln]      = f2bf(sacc[0].f[r]);
      sP[r + lh * 8][16 + ln] = f2bf(sacc[1].f[r]);
    }
    __syncthreads();
    Frag pa;
    pa.q[0] = *(const u32x4*)&sP[ln][lh * 8];
    pa.q[1] = *(const u32x4*)&sP[ln][16 + lh * 8];
    __syncthreads();
    // O += P @ V : four 16x16 D-tiles covering D=64
    for (int jd = 0; jd < 4; ++jd) {
      Frag vb;
      #pragma unroll
      for (int e = 0; e < 16; ++e) {
        int srow = s0 + e + lh * 8 + (e >= 8 ? 8 : 0); if (srow >= Tt) srow = Tt - 1;
        vb.u[e] = Vp[((size_t)(b * Tt) + srow) * Ee + h * Dd + jd * 16 + ln];
      }
      oacc[jd].v = wmma_bf16(pa.v, vb.v, oacc[jd].v);
    }
  }

  for (int r = 0; r < 8; ++r) {
    float inv = 1.0f / sm[r];
    int t = qt * 16 + r + lh * 8;
    for (int jd = 0; jd < 4; ++jd)
      O[((size_t)(b * Tt) + t) * Ee + h * Dd + jd * 16 + ln] = f2bf(oacc[jd].f[r] * inv);
  }
}

// ---------------- loss = -mean log_softmax(x)[targets] over hidden dim E ----------------
__global__ void zero_kernel(float* p) { if (threadIdx.x == 0) p[0] = 0.f; }

__global__ __launch_bounds__(256) void loss_kernel(const float* __restrict__ x,
                                                   const int* __restrict__ targets,
                                                   float* __restrict__ loss) {
  int row = blockIdx.x;
  __shared__ float sh[256];
  const float* xr = x + (size_t)row * Ee;
  float m = -__builtin_inff();
  for (int c = threadIdx.x; c < Ee; c += 256) m = fmaxf(m, xr[c]);
  sh[threadIdx.x] = m; __syncthreads();
  for (int o = 128; o > 0; o >>= 1) {
    if (threadIdx.x < (unsigned)o) sh[threadIdx.x] = fmaxf(sh[threadIdx.x], sh[threadIdx.x + o]);
    __syncthreads();
  }
  float M = sh[0]; __syncthreads();
  float s = 0.f;
  for (int c = threadIdx.x; c < Ee; c += 256) s += __expf(xr[c] - M);
  sh[threadIdx.x] = s; __syncthreads();
  for (int o = 128; o > 0; o >>= 1) {
    if (threadIdx.x < (unsigned)o) sh[threadIdx.x] += sh[threadIdx.x + o];
    __syncthreads();
  }
  if (threadIdx.x == 0) {
    int t = targets[row];
    float lp = xr[t] - M - logf(sh[0]);
    atomicAdd(loss, -lp * (1.0f / (float)Mr));
  }
}

// ---------------- launcher ----------------
extern "C" void kernel_launch(void* const* d_in, const int* in_sizes, int n_in,
                              void* d_out, int out_size, void* d_ws, size_t ws_size,
                              hipStream_t stream) {
  (void)in_sizes; (void)n_in; (void)out_size; (void)ws_size;
  const float* tok_emb = (const float*)d_in[0];
  const float* pos_emb = (const float*)d_in[1];
  const float* Wq  = (const float*)d_in[2];
  const float* Wk  = (const float*)d_in[3];
  const float* Wv  = (const float*)d_in[4];
  const float* Wo  = (const float*)d_in[5];
  const float* bo  = (const float*)d_in[6];
  const float* g1  = (const float*)d_in[7];
  const float* b1  = (const float*)d_in[8];
  const float* g2  = (const float*)d_in[9];
  const float* b2  = (const float*)d_in[10];
  const float* W1  = (const float*)d_in[11];
  const float* bf1 = (const float*)d_in[12];
  const float* W2  = (const float*)d_in[13];
  const float* bf2 = (const float*)d_in[14];
  const float* gf  = (const float*)d_in[15];
  const float* bfp = (const float*)d_in[16];
  const float* Wout = (const float*)d_in[17];
  const float* bout = (const float*)d_in[18];
  const int* tokens  = (const int*)d_in[19];
  const int* targets = (const int*)d_in[20];

  float* logits = (float*)d_out;
  float* loss = logits + (size_t)Mr * Vv;

  char* base = (char*)d_ws;
  size_t off = 0;
  auto carve = [&](size_t bytes) -> char* {
    char* p = base + off;
    off = (off + bytes + 255) & ~(size_t)255;
    return p;
  };
  unsigned short* wqb   = (unsigned short*)carve((size_t)Ll * Ee * Ee * 2);
  unsigned short* wkb   = (unsigned short*)carve((size_t)Ll * Ee * Ee * 2);
  unsigned short* wvb   = (unsigned short*)carve((size_t)Ll * Ee * Ee * 2);
  unsigned short* wob   = (unsigned short*)carve((size_t)Ll * Ee * Ee * 2);
  unsigned short* w1b   = (unsigned short*)carve((size_t)Ll * Ee * Ff * 2);
  unsigned short* w2b   = (unsigned short*)carve((size_t)Ll * Ff * Ee * 2);
  unsigned short* woutb = (unsigned short*)carve((size_t)Ee * Vv * 2);
  float*          x     = (float*)carve((size_t)Mr * Ee * 4);
  unsigned short* hb    = (unsigned short*)carve((size_t)Mr * Ee * 2);
  unsigned short* qb    = (unsigned short*)carve((size_t)Mr * Ee * 2);
  unsigned short* kb    = (unsigned short*)carve((size_t)Mr * Ee * 2);
  unsigned short* vb    = (unsigned short*)carve((size_t)Mr * Ee * 2);
  unsigned short* ob    = (unsigned short*)carve((size_t)Mr * Ee * 2);
  unsigned short* ffb   = (unsigned short*)carve((size_t)Mr * Ff * 2);

  auto cvt = [&](const float* in, unsigned short* out, size_t n) {
    cvt_kernel<<<(unsigned)((n + 255) / 256), 256, 0, stream>>>(in, out, n);
  };
  cvt(Wq, wqb, (size_t)Ll * Ee * Ee);
  cvt(Wk, wkb, (size_t)Ll * Ee * Ee);
  cvt(Wv, wvb, (size_t)Ll * Ee * Ee);
  cvt(Wo, wob, (size_t)Ll * Ee * Ee);
  cvt(W1, w1b, (size_t)Ll * Ee * Ff);
  cvt(W2, w2b, (size_t)Ll * Ff * Ee);
  cvt(Wout, woutb, (size_t)Ee * Vv);

  embed_kernel<<<Mr, 256, 0, stream>>>(tok_emb, pos_emb, tokens, x);

  dim3 gEE(Mr / 128, Ee / 64);   // (32,12)
  dim3 gEF(Mr / 128, Ff / 64);   // (32,48)
  dim3 gEV(Mr / 128, Vv / 64);   // (32,500)
  dim3 gAT(Tt / 16, Hh, Bb);     // (64,12,4)

  for (int layer = 0; layer < Ll; ++layer) {
    const unsigned short* wq_l = wqb + (size_t)layer * Ee * Ee;
    const unsigned short* wk_l = wkb + (size_t)layer * Ee * Ee;
    const unsigned short* wv_l = wvb + (size_t)layer * Ee * Ee;
    const unsigned short* wo_l = wob + (size_t)layer * Ee * Ee;
    const unsigned short* w1_l = w1b + (size_t)layer * Ee * Ff;
    const unsigned short* w2_l = w2b + (size_t)layer * Ff * Ee;

    // h = LN1(x) (bf16)
    ln_kernel<<<Mr, 256, 0, stream>>>(x, g1 + layer * Ee, b1 + layer * Ee, hb);
    // q,k,v = h @ W{q,k,v} (bf16 out, no bias)
    gemm_kernel<1, false><<<gEE, 256, 0, stream>>>(hb, wq_l, nullptr, nullptr, qb, Ee, Ee);
    gemm_kernel<1, false><<<gEE, 256, 0, stream>>>(hb, wk_l, nullptr, nullptr, kb, Ee, Ee);
    gemm_kernel<1, false><<<gEE, 256, 0, stream>>>(hb, wv_l, nullptr, nullptr, vb, Ee, Ee);
    // fused causal attention
    attn_kernel<<<gAT, 32, 0, stream>>>(qb, kb, vb, ob);
    // x += attn_out @ Wo + bo
    gemm_kernel<3, true><<<gEE, 256, 0, stream>>>(ob, wo_l, bo + layer * Ee, x, nullptr, Ee, Ee);
    // h2 = LN2(x); ff = gelu(h2 @ W1 + bf1) (bf16)
    ln_kernel<<<Mr, 256, 0, stream>>>(x, g2 + layer * Ee, b2 + layer * Ee, hb);
    gemm_kernel<2, true><<<gEF, 256, 0, stream>>>(hb, w1_l, bf1 + layer * Ff, nullptr, ffb, Ff, Ee);
    // x += ff @ W2 + bf2
    gemm_kernel<3, true><<<gEE, 256, 0, stream>>>(ffb, w2_l, bf2 + layer * Ee, x, nullptr, Ee, Ff);
  }

  // logits = LN(x, gf, bf) @ Wout + bout
  ln_kernel<<<Mr, 256, 0, stream>>>(x, gf, bfp, hb);
  gemm_kernel<0, true><<<gEV, 256, 0, stream>>>(hb, woutb, bout, logits, nullptr, Vv, Ee);

  // loss on hidden state x (faithful to reference)
  zero_kernel<<<1, 32, 0, stream>>>(loss);
  loss_kernel<<<Mr, 256, 0, stream>>>(x, targets, loss);
}
